// AutoEncoder_59803124629848
// MI455X (gfx1250) — matrix operand order repaired and checked by reading, not compiled, not run on hardware
//
#include <hip/hip_runtime.h>
#include <stdint.h>

typedef __attribute__((ext_vector_type(16))) __bf16 v16bf;
typedef __attribute__((ext_vector_type(8)))  float  v8f;

struct alignas(16) U128 { uint32_t x, y, z, w; };
union Frag { U128 u[2]; v16bf v; };

__device__ __forceinline__ unsigned short f2bf(float f) {
    uint32_t u = __float_as_uint(f);
    uint32_t r = (u + 0x7FFFu + ((u >> 16) & 1u)) >> 16;  // RNE
    return (unsigned short)r;
}

// ---------------- fp32 -> bf16 elementwise ----------------
__global__ void cvt_f32_bf16(const float* __restrict__ in,
                             unsigned short* __restrict__ out, int n) {
    int i = blockIdx.x * blockDim.x + threadIdx.x;
    if (i < n) out[i] = f2bf(in[i]);
}

// ---------------- fused transpose + convert: W[K,N] f32 -> Wt[N,K] bf16 ----
__global__ void transpose_cvt(const float* __restrict__ W,
                              unsigned short* __restrict__ Wt, int K, int N) {
    __shared__ unsigned short tile[32][33];
    int n0 = blockIdx.x * 32, k0 = blockIdx.y * 32;
    int tx = threadIdx.x, ty = threadIdx.y;          // 32 x 8
    for (int j = 0; j < 4; ++j) {
        int k = k0 + ty + j * 8;
        tile[ty + j * 8][tx] = f2bf(W[(size_t)k * N + n0 + tx]);
    }
    __syncthreads();
    for (int j = 0; j < 4; ++j) {
        int n = n0 + ty + j * 8;
        Wt[(size_t)n * K + k0 + tx] = tile[tx][ty + j * 8];
    }
}

// ---------------- bf16 WMMA GEMM: C[M,N] = A[M,K] * Wt[N,K]^T + bias, act --
// ACT: 0=none 1=relu 2=sigmoid
template <int BN, int ACT, bool WF32, bool WBF16>
__global__ __launch_bounds__(256) void gemm_wmma(
    const unsigned short* __restrict__ A, const unsigned short* __restrict__ Bt,
    const float* __restrict__ bias, float* __restrict__ outF,
    unsigned short* __restrict__ outB, int M, int N, int K) {
    constexpr int BM = 128;
    constexpr int WAVES_N = BN / 64;          // 1 or 2
    constexpr int WAVES_M = 8 / WAVES_N;      // 8 or 4
    constexpr int WM = BM / WAVES_M;          // 16 or 32
    constexpr int MT = WM / 16;               // 1 or 2
    constexpr int NT = 4;                     // 64 cols / 16
    constexpr int LS = 40;                    // LDS row stride (bf16 elems)

    __shared__ alignas(16) unsigned short sA[BM * LS];
    __shared__ alignas(16) unsigned short sB[BN * LS];

    const int tid  = threadIdx.x;
    const int lane = tid & 31;
    const int wave = tid >> 5;
    const int waveN = wave % WAVES_N;
    const int waveM = wave / WAVES_N;
    const int rowBase = blockIdx.y * BM;
    const int colBase = blockIdx.x * BN;
    const int wRow = waveM * WM;
    const int wCol = waveN * 64;

    v8f acc[MT][NT];
#pragma unroll
    for (int mi = 0; mi < MT; ++mi)
#pragma unroll
        for (int ni = 0; ni < NT; ++ni)
            acc[mi][ni] = v8f{0.f, 0.f, 0.f, 0.f, 0.f, 0.f, 0.f, 0.f};

    const int arow = tid >> 1;        // 0..127
    const int aseg = tid & 1;         // 0/1 -> 16 bf16 each
    const bool hasB = tid < BN * 2;
    const int brow = tid >> 1;

    const int mLane = lane & 15;             // row within A tile / col within B tile
    const int aK = (lane >> 4) * 8;          // A: K base {0,8}, pairs at +16
    const int bK = (lane >> 4) * 8;          // B: K base {0,8}, pairs at +16

    for (int k0 = 0; k0 < K; k0 += 32) {
        U128 a0, a1, b0{}, b1{};
        {
            const U128* g = (const U128*)(A + (size_t)(rowBase + arow) * K + k0 + aseg * 16);
            a0 = g[0]; a1 = g[1];
        }
        if (hasB) {
            const U128* g = (const U128*)(Bt + (size_t)(colBase + brow) * K + k0 + (tid & 1) * 16);
            b0 = g[0]; b1 = g[1];
        }
        if (k0 + 32 < K)  // gfx1250 global_prefetch_b8 of next K slab
            __builtin_prefetch(A + (size_t)(rowBase + arow) * K + k0 + 32, 0, 1);

        __syncthreads();  // previous iteration's readers done
        {
            U128* s = (U128*)(sA + arow * LS + aseg * 16);
            s[0] = a0; s[1] = a1;
        }
        if (hasB) {
            U128* s = (U128*)(sB + brow * LS + (tid & 1) * 16);
            s[0] = b0; s[1] = b1;
        }
        __syncthreads();

        Frag fa[MT], fb[NT];
#pragma unroll
        for (int mi = 0; mi < MT; ++mi) {
            const unsigned short* p = sA + (wRow + mi * 16 + mLane) * LS + aK;
            fa[mi].u[0] = *(const U128*)p;          // K = aK .. aK+7
            fa[mi].u[1] = *(const U128*)(p + 16);   // K = aK+16 .. aK+23
        }
#pragma unroll
        for (int ni = 0; ni < NT; ++ni) {
            const unsigned short* p = sB + (wCol + ni * 16 + mLane) * LS + bK;
            fb[ni].u[0] = *(const U128*)p;          // K = bK .. bK+7
            fb[ni].u[1] = *(const U128*)(p + 16);   // K = bK+16 .. bK+23
        }
#pragma unroll
        for (int mi = 0; mi < MT; ++mi)
#pragma unroll
            for (int ni = 0; ni < NT; ++ni)
                acc[mi][ni] = __builtin_amdgcn_wmma_f32_16x16x32_bf16(
                    false, fa[mi].v, false, fb[ni].v, (short)0, acc[mi][ni],
                    false, false);
    }

    // Epilogue: C/D layout -> M = r + 8*(lane>>4), N = lane&15
    const int rq = lane >> 4;
    const int cq = lane & 15;
#pragma unroll
    for (int mi = 0; mi < MT; ++mi) {
#pragma unroll
        for (int ni = 0; ni < NT; ++ni) {
            const int gc = colBase + wCol + ni * 16 + cq;
            const float bv = bias[gc];
#pragma unroll
            for (int r = 0; r < 8; ++r) {
                const int gr = rowBase + wRow + mi * 16 + rq * 8 + r;
                float v = acc[mi][ni][r] + bv;
                if constexpr (ACT == 1) v = fmaxf(v, 0.0f);
                if constexpr (ACT == 2) v = 1.0f / (1.0f + __expf(-v));
                const size_t idx = (size_t)gr * N + gc;
                if constexpr (WF32)  outF[idx] = v;
                if constexpr (WBF16) outB[idx] = f2bf(v);
            }
        }
    }
    (void)M;
}

extern "C" void kernel_launch(void* const* d_in, const int* in_sizes, int n_in,
                              void* d_out, int out_size, void* d_ws, size_t ws_size,
                              hipStream_t stream) {
    (void)in_sizes; (void)n_in; (void)out_size; (void)ws_size;
    constexpr int Bm = 8192, DIN = 1024, DC = 64, DST = 1024, H = 2048;

    const float* x   = (const float*)d_in[0];
    const float* We1 = (const float*)d_in[1];
    const float* be1 = (const float*)d_in[2];
    const float* We2 = (const float*)d_in[3];
    const float* be2 = (const float*)d_in[4];
    const float* We3 = (const float*)d_in[5];
    const float* be3 = (const float*)d_in[6];
    const float* Wd1 = (const float*)d_in[7];
    const float* bd1 = (const float*)d_in[8];
    const float* Wd2 = (const float*)d_in[9];
    const float* bd2 = (const float*)d_in[10];
    const float* Wd3 = (const float*)d_in[11];
    const float* bd3 = (const float*)d_in[12];
    float* out = (float*)d_out;   // [B*64] control, then [B*1024] state

    char* p = (char*)d_ws;
    unsigned short* xb   = (unsigned short*)p; p += (size_t)Bm * DIN * 2;
    unsigned short* We1t = (unsigned short*)p; p += (size_t)H * DIN * 2;
    unsigned short* We2t = (unsigned short*)p; p += (size_t)H * H * 2;
    unsigned short* We3t = (unsigned short*)p; p += (size_t)DC * H * 2;
    unsigned short* Wd1t = (unsigned short*)p; p += (size_t)H * DC * 2;
    unsigned short* Wd2t = (unsigned short*)p; p += (size_t)H * H * 2;
    unsigned short* Wd3t = (unsigned short*)p; p += (size_t)DST * H * 2;
    unsigned short* h1   = (unsigned short*)p; p += (size_t)Bm * H * 2;   // also d1
    unsigned short* h2   = (unsigned short*)p; p += (size_t)Bm * H * 2;   // also d2
    unsigned short* cb   = (unsigned short*)p; p += (size_t)Bm * DC * 2;

    // --- convert x to bf16 ---
    {
        int n = Bm * DIN;
        cvt_f32_bf16<<<(n + 255) / 256, 256, 0, stream>>>(x, xb, n);
    }
    // --- transpose+convert weights: W[K,N] -> Wt[N,K] bf16 ---
    dim3 tb(32, 8);
    transpose_cvt<<<dim3(H   / 32, DIN / 32), tb, 0, stream>>>(We1, We1t, DIN, H);
    transpose_cvt<<<dim3(H   / 32, H   / 32), tb, 0, stream>>>(We2, We2t, H,   H);
    transpose_cvt<<<dim3(DC  / 32, H   / 32), tb, 0, stream>>>(We3, We3t, H,   DC);
    transpose_cvt<<<dim3(H   / 32, DC  / 32), tb, 0, stream>>>(Wd1, Wd1t, DC,  H);
    transpose_cvt<<<dim3(H   / 32, H   / 32), tb, 0, stream>>>(Wd2, Wd2t, H,   H);
    transpose_cvt<<<dim3(DST / 32, H   / 32), tb, 0, stream>>>(Wd3, Wd3t, H,   DST);

    // --- encoder ---
    gemm_wmma<128, 1, false, true><<<dim3(H / 128, Bm / 128), 256, 0, stream>>>(
        xb, We1t, be1, nullptr, h1, Bm, H, DIN);
    gemm_wmma<128, 1, false, true><<<dim3(H / 128, Bm / 128), 256, 0, stream>>>(
        h1, We2t, be2, nullptr, h2, Bm, H, H);
    gemm_wmma<64, 2, true, true><<<dim3(DC / 64, Bm / 128), 256, 0, stream>>>(
        h2, We3t, be3, out, cb, Bm, DC, H);                 // control (sigmoid)
    // --- decoder ---
    gemm_wmma<128, 1, false, true><<<dim3(H / 128, Bm / 128), 256, 0, stream>>>(
        cb, Wd1t, bd1, nullptr, h1, Bm, H, DC);
    gemm_wmma<128, 1, false, true><<<dim3(H / 128, Bm / 128), 256, 0, stream>>>(
        h1, Wd2t, bd2, nullptr, h2, Bm, H, H);
    gemm_wmma<128, 0, true, false><<<dim3(DST / 128, Bm / 128), 256, 0, stream>>>(
        h2, Wd3t, bd3, out + (size_t)Bm * DC, nullptr, Bm, DST, H);  // state
}
